// AttentionDrop_3238405341579
// MI455X (gfx1250) — matrix operand order repaired and checked by reading, not compiled, not run on hardware
//
#include <hip/hip_runtime.h>
#include <hip/hip_bf16.h>

typedef __attribute__((ext_vector_type(16))) __bf16 v16bf;
typedef __attribute__((ext_vector_type(8)))  float  v8f;
typedef __attribute__((ext_vector_type(4)))  unsigned int u32x4;

#define DIM      1024
#define NSEQ     2048
#define BATCH    2
#define HEADS    16
#define HEAD_DIM 64
// softmax computed in log2 domain: S * (1/sqrt(64)) * log2(e)
#define SCALE_LOG2E 0.18033688f

// ---------- helpers ----------

__device__ __forceinline__ unsigned short f2bf(float f) {
    unsigned int u = __float_as_uint(f);
    unsigned int r = (u + 0x7FFFu + ((u >> 16) & 1u)) >> 16;
    return (unsigned short)r;
}

// packed pair conversion: returns {bf16(a) in [15:0], bf16(b) in [31:16]}
__device__ __forceinline__ unsigned int cvt_pk_bf16(float a, float b) {
#if __has_builtin(__builtin_amdgcn_cvt_pk_bf16_f32)
    auto pk = __builtin_amdgcn_cvt_pk_bf16_f32(a, b);
    union { decltype(pk) p; unsigned int u; } cv;
    cv.p = pk;
    return cv.u;
#else
    return (unsigned int)f2bf(a) | ((unsigned int)f2bf(b) << 16);
#endif
}

union V16U {
    struct { u32x4 a, b; } u;
    v16bf v;
};

// A-operand 16x32 bf16 tile. p = &src[row_of_this_lane * ld + k0] (16B aligned).
// elems 0..15 : K = (hi?16:0) + 0..15
__device__ __forceinline__ v16bf load_a_tile(const unsigned short* p, bool hi) {
    V16U t;
    t.u.a = *(const u32x4*)(p + (hi ? 16 : 0));
    t.u.b = *(const u32x4*)(p + (hi ? 24 : 8));
    return t.v;
}

// B-operand 32x16 bf16 tile. p = &src[col_of_this_lane * ld + k0 + (hi?16:0)]
__device__ __forceinline__ v16bf load_b_tile(const unsigned short* p) {
    V16U t;
    t.u.a = *(const u32x4*)(p);
    t.u.b = *(const u32x4*)(p + 8);
    return t.v;
}

__device__ __forceinline__ v16bf ones_bf16() {
    u32x4 c = {0x3F803F80u, 0x3F803F80u, 0x3F803F80u, 0x3F803F80u};
    V16U t;
    t.u.a = c;
    t.u.b = c;
    return t.v;
}

__device__ __forceinline__ v8f wmma_bf16(v16bf a, v16bf b, v8f c) {
    return __builtin_amdgcn_wmma_f32_16x16x32_bf16(
        false, a, false, b, (short)0, c, false, false);
}

__device__ __forceinline__ float hmax16(float v) {
    v = fmaxf(v, __shfl_xor(v, 1, 32));
    v = fmaxf(v, __shfl_xor(v, 2, 32));
    v = fmaxf(v, __shfl_xor(v, 4, 32));
    v = fmaxf(v, __shfl_xor(v, 8, 32));
    return v;
}

// ---------- pass 0: fp32 -> bf16 (4-wide) ----------

__global__ void cvt_bf16_kernel(const float* __restrict__ in,
                                unsigned short* __restrict__ out, int n) {
    int i = (blockIdx.x * blockDim.x + threadIdx.x) * 4;
    if (i < n) {
        float4 f = *(const float4*)(in + i);
        uint2 v;
        v.x = cvt_pk_bf16(f.x, f.y);
        v.y = cvt_pk_bf16(f.z, f.w);
        *(uint2*)(out + i) = v;
    }
}

// ---------- pass 1: fused QKV projection (32 rows x 16 cols x 3 mats per wave) ----------
// x:[B*N, C] bf16 ; W:[C, C] bf16. Q,K -> [B,H,N,D] ; V -> transposed [B,H,D,N]
__global__ __launch_bounds__(256)
void qkv_kernel(const unsigned short* __restrict__ xb,
                const unsigned short* __restrict__ wq,
                const unsigned short* __restrict__ wk,
                const unsigned short* __restrict__ wv,
                unsigned short* __restrict__ Q,
                unsigned short* __restrict__ K,
                unsigned short* __restrict__ Vt) {
    int tid  = threadIdx.x;
    int lane = tid & 31;
    bool hi  = lane >= 16;
    int l15  = lane & 15;
    int w    = blockIdx.x * 8 + (tid >> 5);     // 8192 waves total
    int tr   = w >> 6;                          // 0..127 (32-row tiles)
    int tc   = w & 63;                          // 0..63  (16-col tiles)

    const unsigned short* a0_base = xb + (size_t)(tr * 32 + l15) * DIM;
    const unsigned short* a1_base = xb + (size_t)(tr * 32 + 16 + l15) * DIM;
    const unsigned short* bq_base = wq + (size_t)(tc * 16 + l15) * DIM + (hi ? 16 : 0);
    const unsigned short* bk_base = wk + (size_t)(tc * 16 + l15) * DIM + (hi ? 16 : 0);
    const unsigned short* bv_base = wv + (size_t)(tc * 16 + l15) * DIM + (hi ? 16 : 0);

    v8f cq[2] = {{}, {}}, ck[2] = {{}, {}}, cv[2] = {{}, {}};
    for (int k0 = 0; k0 < DIM; k0 += 32) {
        v16bf a0 = load_a_tile(a0_base + k0, hi);
        v16bf a1 = load_a_tile(a1_base + k0, hi);
        v16bf bq = load_b_tile(bq_base + k0);
        cq[0] = wmma_bf16(a0, bq, cq[0]);
        cq[1] = wmma_bf16(a1, bq, cq[1]);
        v16bf bk = load_b_tile(bk_base + k0);
        ck[0] = wmma_bf16(a0, bk, ck[0]);
        ck[1] = wmma_bf16(a1, bk, ck[1]);
        v16bf bv = load_b_tile(bv_base + k0);
        cv[0] = wmma_bf16(a0, bv, cv[0]);
        cv[1] = wmma_bf16(a1, bv, cv[1]);
    }

    int j = tc * 16 + l15;          // output channel
    int h = j >> 6;
    int d = j & 63;
#pragma unroll
    for (int s = 0; s < 2; ++s) {
#pragma unroll
        for (int r = 0; r < 8; ++r) {
            int g = tr * 32 + s * 16 + (hi ? 8 : 0) + r;   // global row in [0, B*N)
            int b = g >> 11;
            int n = g & (NSEQ - 1);
            size_t qi = (((size_t)(b * HEADS + h) * NSEQ) + n) * HEAD_DIM + d;
            Q[qi] = f2bf(cq[s][r]);
            K[qi] = f2bf(ck[s][r]);
            Vt[(((size_t)(b * HEADS + h) * HEAD_DIM) + d) * NSEQ + n] = f2bf(cv[s][r]);
        }
    }
}

// ---------- pass 2: flash attention ----------
// 8 waves/block; each wave owns 32 query rows (two 16-row WMMA tiles) -> 256 rows/block.
// K/V operand tiles shared by both query tiles. Row sums of P computed with an
// extra WMMA against an all-ones B operand (replaces shuffle-tree reductions).
__global__ __launch_bounds__(256)
void attn_kernel(const unsigned short* __restrict__ Q,
                 const unsigned short* __restrict__ K,
                 const unsigned short* __restrict__ Vt,
                 unsigned short* __restrict__ ctx) {
    __shared__ __align__(16) unsigned short plds[8 * 1024];  // 2x (16x32) bf16 per wave

    int tid  = threadIdx.x;
    int lane = tid & 31;
    int wv   = tid >> 5;
    bool hi  = lane >= 16;
    int l15  = lane & 15;

    int blk = blockIdx.x;            // b*128 + h*8 + qb
    int qb  = blk & 7;
    int h   = (blk >> 3) & 15;
    int b   = blk >> 7;
    int n0  = qb * 256 + wv * 32;

    const unsigned short* Qh = Q  + (size_t)(b * HEADS + h) * NSEQ * HEAD_DIM;
    const unsigned short* Kh = K  + (size_t)(b * HEADS + h) * NSEQ * HEAD_DIM;
    const unsigned short* Vh = Vt + (size_t)(b * HEADS + h) * HEAD_DIM * NSEQ;
    unsigned short* myp = plds + wv * 1024;

    const v16bf bones = ones_bf16();

    // Q A-operand tiles: 2 query tiles x 2 K-chunks, resident for the whole loop.
    v16bf aq[2][2];
#pragma unroll
    for (int q = 0; q < 2; ++q) {
#pragma unroll
        for (int c = 0; c < 2; ++c)
            aq[q][c] = load_a_tile(Qh + (size_t)(n0 + q * 16 + l15) * HEAD_DIM + c * 32, hi);
    }

    float mrow[2][8], lrow[2][8];
    v8f o[2][4] = {{{}, {}, {}, {}}, {{}, {}, {}, {}}};
#pragma unroll
    for (int q = 0; q < 2; ++q)
#pragma unroll
        for (int r = 0; r < 8; ++r) { mrow[q][r] = -3.0e38f; lrow[q][r] = 0.0f; }

    for (int m0 = 0; m0 < NSEQ; m0 += 32) {
        // K B-tiles loaded once, reused by both query tiles.
        v16bf bk[2][2];
#pragma unroll
        for (int ct = 0; ct < 2; ++ct)
#pragma unroll
            for (int kc = 0; kc < 2; ++kc)
                bk[ct][kc] = load_b_tile(Kh + (size_t)(m0 + ct * 16 + l15) * HEAD_DIM
                                            + kc * 32 + (hi ? 16 : 0));

#pragma unroll
        for (int q = 0; q < 2; ++q) {
            // ---- S tile 16x32 ----
            v8f s0 = {}, s1 = {};
            s0 = wmma_bf16(aq[q][0], bk[0][0], s0);
            s0 = wmma_bf16(aq[q][1], bk[0][1], s0);
            s1 = wmma_bf16(aq[q][0], bk[1][0], s1);
            s1 = wmma_bf16(aq[q][1], bk[1][1], s1);

            // ---- online softmax in log2 domain (row sums deferred to WMMA) ----
            float alpha[8];
#pragma unroll
            for (int r = 0; r < 8; ++r) {
                float v0 = s0[r] * SCALE_LOG2E;
                float v1 = s1[r] * SCALE_LOG2E;
                float t  = hmax16(fmaxf(v0, v1));
                float mn = fmaxf(mrow[q][r], t);
                float a  = exp2f(mrow[q][r] - mn);
                lrow[q][r] *= a;
                mrow[q][r]  = mn;
                alpha[r]    = a;
                s0[r] = exp2f(v0 - mn);
                s1[r] = exp2f(v1 - mn);
            }
#pragma unroll
            for (int c = 0; c < 4; ++c)
#pragma unroll
                for (int r = 0; r < 8; ++r) o[q][c][r] *= alpha[r];

            // ---- P (C-layout) -> LDS, packed pair conversion ----
            int M = hi ? 8 : 0;
#pragma unroll
            for (int r = 0; r < 8; ++r) {
                unsigned int pk = cvt_pk_bf16(s0[r], s1[r]);
                myp[q * 512 + (M + r) * 32 + l15]      = (unsigned short)(pk & 0xFFFFu);
                myp[q * 512 + (M + r) * 32 + 16 + l15] = (unsigned short)(pk >> 16);
            }
        }
        asm volatile("s_wait_dscnt 0" ::: "memory");

        v16bf ap[2];
#pragma unroll
        for (int q = 0; q < 2; ++q)
            ap[q] = load_a_tile(myp + q * 512 + l15 * 32, hi);

        // ---- row sums of P via WMMA against all-ones B ----
#pragma unroll
        for (int q = 0; q < 2; ++q) {
            v8f rs = {};
            rs = wmma_bf16(ap[q], bones, rs);
#pragma unroll
            for (int r = 0; r < 8; ++r) lrow[q][r] += rs[r];
        }

        // ---- O += P . V ; V B-tiles shared by both query tiles ----
#pragma unroll
        for (int c = 0; c < 4; ++c) {
            v16bf bv = load_b_tile(Vh + (size_t)(c * 16 + l15) * NSEQ + m0 + (hi ? 16 : 0));
            o[0][c] = wmma_bf16(ap[0], bv, o[0][c]);
            o[1][c] = wmma_bf16(ap[1], bv, o[1][c]);
        }
    }

    // ---- normalize and store context [B,N,C] bf16 ----
#pragma unroll
    for (int q = 0; q < 2; ++q) {
#pragma unroll
        for (int r = 0; r < 8; ++r) {
            float inv = 1.0f / lrow[q][r];
            int row = n0 + q * 16 + (hi ? 8 : 0) + r;
            size_t g = (size_t)b * NSEQ + row;
            unsigned int p01 = cvt_pk_bf16(o[q][0][r] * inv, o[q][1][r] * inv);
            unsigned int p23 = cvt_pk_bf16(o[q][2][r] * inv, o[q][3][r] * inv);
            unsigned short* cp = ctx + g * DIM + h * HEAD_DIM + l15;
            cp[0]  = (unsigned short)(p01 & 0xFFFFu);
            cp[16] = (unsigned short)(p01 >> 16);
            cp[32] = (unsigned short)(p23 & 0xFFFFu);
            cp[48] = (unsigned short)(p23 >> 16);
        }
    }
}

// ---------- pass 3: output projection + bias (64 rows x 32 cols per wave) ----------

__global__ __launch_bounds__(256)
void proj_kernel(const unsigned short* __restrict__ ctx,
                 const unsigned short* __restrict__ wo,
                 const float* __restrict__ bo,
                 float* __restrict__ out) {
    int tid  = threadIdx.x;
    int lane = tid & 31;
    bool hi  = lane >= 16;
    int l15  = lane & 15;
    int w    = blockIdx.x * 8 + (tid >> 5);     // 2048 waves total
    int tr   = w >> 5;                          // 0..63 (64-row tiles)
    int tc   = w & 31;                          // 0..31 (32-col tiles)

    const unsigned short* a_base[4];
#pragma unroll
    for (int s = 0; s < 4; ++s)
        a_base[s] = ctx + (size_t)(tr * 64 + s * 16 + l15) * DIM;
    const unsigned short* b_base[2];
#pragma unroll
    for (int ct = 0; ct < 2; ++ct)
        b_base[ct] = wo + (size_t)(tc * 32 + ct * 16 + l15) * DIM + (hi ? 16 : 0);

    v8f acc[4][2] = {{{}, {}}, {{}, {}}, {{}, {}}, {{}, {}}};
    for (int k0 = 0; k0 < DIM; k0 += 32) {
        v16bf a[4];
#pragma unroll
        for (int s = 0; s < 4; ++s) a[s] = load_a_tile(a_base[s] + k0, hi);
#pragma unroll
        for (int ct = 0; ct < 2; ++ct) {
            v16bf bw = load_b_tile(b_base[ct] + k0);
#pragma unroll
            for (int s = 0; s < 4; ++s) acc[s][ct] = wmma_bf16(a[s], bw, acc[s][ct]);
        }
    }

#pragma unroll
    for (int ct = 0; ct < 2; ++ct) {
        int j = tc * 32 + ct * 16 + l15;
        float bias = bo[j];
#pragma unroll
        for (int s = 0; s < 4; ++s) {
#pragma unroll
            for (int r = 0; r < 8; ++r) {
                int g = tr * 64 + s * 16 + (hi ? 8 : 0) + r;
                out[(size_t)g * DIM + j] = acc[s][ct][r] + bias;
            }
        }
    }
}

// ---------- launch ----------

extern "C" void kernel_launch(void* const* d_in, const int* in_sizes, int n_in,
                              void* d_out, int out_size, void* d_ws, size_t ws_size,
                              hipStream_t stream) {
    const float* x  = (const float*)d_in[0];
    const float* Wq = (const float*)d_in[1];
    const float* Wk = (const float*)d_in[2];
    const float* Wv = (const float*)d_in[3];
    const float* Wo = (const float*)d_in[4];
    const float* bo = (const float*)d_in[5];
    float* out = (float*)d_out;

    char* ws = (char*)d_ws;
    const size_t MB = 1u << 20;
    unsigned short* xb  = (unsigned short*)(ws + 0 * MB);   // 8 MB  [B*N, C]
    unsigned short* wqb = (unsigned short*)(ws + 8 * MB);   // 2 MB
    unsigned short* wkb = (unsigned short*)(ws + 10 * MB);  // 2 MB
    unsigned short* wvb = (unsigned short*)(ws + 12 * MB);  // 2 MB
    unsigned short* wob = (unsigned short*)(ws + 14 * MB);  // 2 MB
    unsigned short* Qb  = (unsigned short*)(ws + 16 * MB);  // 8 MB  [B,H,N,D]
    unsigned short* Kb  = (unsigned short*)(ws + 24 * MB);  // 8 MB  [B,H,N,D]
    unsigned short* Vt  = (unsigned short*)(ws + 32 * MB);  // 8 MB  [B,H,D,N]
    unsigned short* ctx = (unsigned short*)(ws + 40 * MB);  // 8 MB  [B,N,C]

    const int nX = BATCH * NSEQ * DIM;     // 4,194,304
    const int nW = DIM * DIM;              // 1,048,576

    cvt_bf16_kernel<<<(nX / 4 + 255) / 256, 256, 0, stream>>>(x,  xb,  nX);
    cvt_bf16_kernel<<<(nW / 4 + 255) / 256, 256, 0, stream>>>(Wq, wqb, nW);
    cvt_bf16_kernel<<<(nW / 4 + 255) / 256, 256, 0, stream>>>(Wk, wkb, nW);
    cvt_bf16_kernel<<<(nW / 4 + 255) / 256, 256, 0, stream>>>(Wv, wvb, nW);
    cvt_bf16_kernel<<<(nW / 4 + 255) / 256, 256, 0, stream>>>(Wo, wob, nW);

    // 128 row tiles x 64 col tiles = 8192 waves / 8 per block
    qkv_kernel<<<1024, 256, 0, stream>>>(xb, wqb, wkb, wvb, Qb, Kb, Vt);

    // B * H * (N/256) blocks, 8 waves x 32 query rows each
    attn_kernel<<<BATCH * HEADS * (NSEQ / 256), 256, 0, stream>>>(Qb, Kb, Vt, ctx);

    // 64 row tiles x 32 col tiles = 2048 waves / 8 per block
    proj_kernel<<<256, 256, 0, stream>>>(ctx, wob, bo, out);
}
